// PointWiseDecoder1D_35218731828043
// MI455X (gfx1250) — compile-verified, matrix operand order checked by reading, not run on hardware
//
#include <hip/hip_runtime.h>
#include <hip/hip_bf16.h>

// ---------------------------------------------------------------------------
// CDNA5 (gfx1250) implementation of PointWiseDecoder1D forward.
// wave32, WMMA f32_16x16x32_f16 for all GEMMs, LDS Stockham FFT for Hyena.
// ---------------------------------------------------------------------------

typedef _Float16 v16h __attribute__((ext_vector_type(16)));
typedef float    v8f  __attribute__((ext_vector_type(8)));

__device__ __forceinline__ float gelu_f(float x) {
    return 0.5f * x * (1.0f + erff(x * 0.70710678118654752f));
}

template<bool H> struct ElT            { using T = float;    };
template<>       struct ElT<true>      { using T = _Float16; };

union FragU { v16h v; unsigned u[8]; };

// ---------------------------------------------------------------------------
// Generic batched WMMA GEMM: C = act(scale * A(x)B + bias) [+ res], f16 sidecar
//   A: [M,K] (or [K,M] if TRANSA), f32 or f16.  B: [K,N], f32 or f16.
//   batch z: off = (z/H)*s1 + (z%H)*s2 for A/B/C independently.
// Tile: BM=128, BN=128, BK=32; 256 threads = 8 waves, each wave 32x64.
// ---------------------------------------------------------------------------
template<bool TRANSA, bool AH, bool BH, int ACT, bool RES, bool WH>
__global__ __launch_bounds__(256)
void k_gemm(const void* __restrict__ Ap, const void* __restrict__ Bp,
            const float* __restrict__ bias,
            float* __restrict__ C, _Float16* __restrict__ Ch,
            const float* __restrict__ res,
            int M, int N, int K, int lda, int ldb, int ldc,
            float scale, int H,
            long long sA1, long long sA2,
            long long sB1, long long sB2,
            long long sC1, long long sC2)
{
    using AT = typename ElT<AH>::T;
    using BT = typename ElT<BH>::T;

    __shared__ _Float16 As[128][32];
    __shared__ _Float16 Bs[128][32];

    const int tid  = threadIdx.x;
    const int z    = blockIdx.z;
    const long long offA = (long long)(z / H) * sA1 + (long long)(z % H) * sA2;
    const long long offB = (long long)(z / H) * sB1 + (long long)(z % H) * sB2;
    const long long offC = (long long)(z / H) * sC1 + (long long)(z % H) * sC2;

    const AT* __restrict__ A = ((const AT*)Ap) + offA;
    const BT* __restrict__ B = ((const BT*)Bp) + offB;

    const int m0 = blockIdx.y * 128;
    const int n0 = blockIdx.x * 128;

    const int lane = tid & 31;
    const int wave = tid >> 5;
    const int wm   = wave >> 1;        // 0..3 -> 32-row slab
    const int wn   = wave & 1;         // 0..1 -> 64-col slab
    const int lm   = lane & 15;
    const int lh   = lane >> 4;

    // no outstanding tensor ops: explicit gfx1250 wait (no-op, keeps TDM path hot)
    __builtin_amdgcn_s_wait_tensorcnt(0);

    v8f acc[2][4];
    const v8f z8 = {0.f, 0.f, 0.f, 0.f, 0.f, 0.f, 0.f, 0.f};
    for (int a = 0; a < 2; ++a)
        for (int b = 0; b < 4; ++b) acc[a][b] = z8;

    for (int k0 = 0; k0 < K; k0 += 32) {
        // ---- stage A tile ----
        if (TRANSA) {
            for (int e = tid; e < 4096; e += 256) {
                int kk = e >> 7, m = e & 127;
                float v = (float)A[(long long)(k0 + kk) * lda + (m0 + m)];
                As[m][kk] = (_Float16)v;
            }
        } else {
            for (int e = tid; e < 4096; e += 256) {
                int m = e >> 5, kk = e & 31;
                float v = (float)A[(long long)(m0 + m) * lda + (k0 + kk)];
                As[m][kk] = (_Float16)v;
            }
        }
        // ---- stage B tile ----
        for (int e = tid; e < 4096; e += 256) {
            int kk = e >> 7, n = e & 127;
            int ng = n0 + n;
            float v = (ng < N) ? (float)B[(long long)(k0 + kk) * ldb + ng] : 0.0f;
            Bs[n][kk] = (_Float16)v;
        }
        if (k0 + 32 < K) {
            // speculative prefetch of the next tiles (global_prefetch_b8)
            if (TRANSA) __builtin_prefetch(&A[(long long)(k0 + 32 + (tid >> 7)) * lda + m0 + (tid & 127)], 0, 1);
            else        __builtin_prefetch(&A[(long long)(m0 + (tid >> 1)) * lda + k0 + 32], 0, 1);
            __builtin_prefetch(&B[(long long)(k0 + 32 + (tid >> 7)) * ldb + n0 + (tid & 127)], 0, 1);
        }
        __syncthreads();

        // ---- fragments per documented 16x32 f16 layout ----
        FragU afr[2], bfr[4];
        for (int tm = 0; tm < 2; ++tm) {
            int m = wm * 32 + tm * 16 + lm;
            #pragma unroll
            for (int p = 0; p < 8; ++p) {
                int kk0 = ((p >> 2) << 4) + (lh << 3) + ((p & 3) << 1);
                afr[tm].u[p] = *(const unsigned*)&As[m][kk0];
            }
        }
        for (int tn = 0; tn < 4; ++tn) {
            int n = wn * 64 + tn * 16 + lm;
            #pragma unroll
            for (int p = 0; p < 8; ++p) {
                int kk0 = ((p >> 2) << 4) + (lh << 3) + ((p & 3) << 1);
                bfr[tn].u[p] = *(const unsigned*)&Bs[n][kk0];
            }
        }
        #pragma unroll
        for (int tm = 0; tm < 2; ++tm)
            #pragma unroll
            for (int tn = 0; tn < 4; ++tn)
                acc[tm][tn] = __builtin_amdgcn_wmma_f32_16x16x32_f16(
                    false, afr[tm].v, false, bfr[tn].v,
                    (short)0, acc[tm][tn], false, false);
        __syncthreads();
    }

    // ---- epilogue ----
    for (int tm = 0; tm < 2; ++tm) {
        for (int tn = 0; tn < 4; ++tn) {
            #pragma unroll
            for (int r = 0; r < 8; ++r) {
                int m  = m0 + wm * 32 + tm * 16 + lh * 8 + r;
                int ng = n0 + wn * 64 + tn * 16 + lm;
                if (ng < N) {
                    float v = acc[tm][tn][r] * scale;
                    if (bias) v += bias[ng];
                    if (ACT == 1) v = gelu_f(v);
                    long long idx = offC + (long long)m * ldc + ng;
                    if (RES) v += res[idx];
                    C[idx] = v;
                    if (WH) Ch[idx] = (_Float16)v;
                }
            }
        }
    }
}

// ---------------------------------------------------------------------------
// Row layernorm over 128 features, optional residual add.
// ---------------------------------------------------------------------------
template<bool ADD>
__global__ __launch_bounds__(128)
void k_layernorm(const float* __restrict__ in, const float* __restrict__ g,
                 const float* __restrict__ b, const float* __restrict__ res,
                 float* __restrict__ out)
{
    __shared__ float red[128];
    const int tid = threadIdx.x;
    const long long row = blockIdx.x;
    float v = in[row * 128 + tid];

    red[tid] = v; __syncthreads();
    for (int s = 64; s > 0; s >>= 1) { if (tid < s) red[tid] += red[tid + s]; __syncthreads(); }
    float mu = red[0] * (1.0f / 128.0f);
    __syncthreads();
    float d = v - mu;
    red[tid] = d * d; __syncthreads();
    for (int s = 64; s > 0; s >>= 1) { if (tid < s) red[tid] += red[tid + s]; __syncthreads(); }
    float rstd = rsqrtf(red[0] * (1.0f / 128.0f) + 1e-5f);

    float o = d * rstd * g[tid] + b[tid];
    if (ADD) o += res[row * 128 + tid];
    out[row * 128 + tid] = o;
}

// ---------------------------------------------------------------------------
// Instance-norm over sequence + rotary, writes f16 [b*h, 2048, 128].
// grid = (64 j, 8 h, 8 b), block = 256.
// ---------------------------------------------------------------------------
__global__ __launch_bounds__(256)
void k_instrot(const float* __restrict__ qf, const float* __restrict__ pos,
               _Float16* __restrict__ qr)
{
    __shared__ float r0[256], r1[256], r2[256], r3[256];
    const int tid = threadIdx.x;
    const int j = blockIdx.x, h = blockIdx.y, b = blockIdx.z;
    const int c0 = h * 128 + j, c1 = c0 + 64;

    float s0 = 0.f, ss0 = 0.f, s1 = 0.f, ss1 = 0.f;
    for (int i = tid; i < 2048; i += 256) {
        float a = qf[((long long)(b * 2048 + i)) * 1024 + c0];
        float c = qf[((long long)(b * 2048 + i)) * 1024 + c1];
        s0 += a; ss0 += a * a; s1 += c; ss1 += c * c;
    }
    r0[tid] = s0; r1[tid] = ss0; r2[tid] = s1; r3[tid] = ss1;
    __syncthreads();
    for (int s = 128; s > 0; s >>= 1) {
        if (tid < s) { r0[tid] += r0[tid+s]; r1[tid] += r1[tid+s];
                       r2[tid] += r2[tid+s]; r3[tid] += r3[tid+s]; }
        __syncthreads();
    }
    const float inv_n = 1.0f / 2048.0f;
    float m0 = r0[0] * inv_n, m1 = r2[0] * inv_n;
    float rs0 = rsqrtf(fmaxf(r1[0] * inv_n - m0 * m0, 0.f) + 1e-5f);
    float rs1 = rsqrtf(fmaxf(r3[0] * inv_n - m1 * m1, 0.f) + 1e-5f);

    const float invj = __expf(-(2.0f * (float)j / 128.0f) * 9.210340371976184f); // /10000^(2j/d)
    for (int i = tid; i < 2048; i += 256) {
        float a = qf[((long long)(b * 2048 + i)) * 1024 + c0];
        float c = qf[((long long)(b * 2048 + i)) * 1024 + c1];
        float q0 = (a - m0) * rs0, q1 = (c - m1) * rs1;
        float t  = pos[b * 2048 + i] * 2048.0f;
        float fr = t * invj, sn, cs;
        __sincosf(fr, &sn, &cs);
        long long base = ((long long)(b * 8 + h) * 2048 + i) * 128;
        qr[base + j]      = (_Float16)(q0 * cs - q1 * sn);
        qr[base + j + 64] = (_Float16)(q1 * cs + q0 * sn);
    }
}

// ---------------------------------------------------------------------------
// Gaussian Fourier features: feats[m, c] = gelu(sin/cos(2*pi*pos[m]*B[c%128]))
// ---------------------------------------------------------------------------
__global__ __launch_bounds__(256)
void k_feats(const float* __restrict__ pos, const float* __restrict__ Bv,
             float* __restrict__ feats)
{
    const int c = threadIdx.x;
    const long long m = blockIdx.x;
    float w = 6.283185307179586f * pos[m] * Bv[c & 127];
    float s = (c < 128) ? sinf(w) : cosf(w);
    feats[m * 256 + c] = gelu_f(s);
}

// ---------------------------------------------------------------------------
// Hyena depthwise causal conv (k=3): u [16384,384] -> vx=v*x1, x0 in [b,d,t].
// grid = (2048 t, 8 b), block = 128 (d).
// ---------------------------------------------------------------------------
__global__ __launch_bounds__(128)
void k_shortconv(const float* __restrict__ u, const float* __restrict__ sfw,
                 const float* __restrict__ sfb,
                 float* __restrict__ vx, float* __restrict__ x0a)
{
    const int t = blockIdx.x, b = blockIdx.y, d = threadIdx.x;
    const long long r  = (long long)b * 2048 + t;
    float uc[3];
    #pragma unroll
    for (int s = 0; s < 3; ++s) {
        int c = d + s * 128;
        float um2 = (t >= 2) ? u[(r - 2) * 384 + c] : 0.f;
        float um1 = (t >= 1) ? u[(r - 1) * 384 + c] : 0.f;
        float u0  = u[r * 384 + c];
        uc[s] = sfw[c * 3 + 0] * um2 + sfw[c * 3 + 1] * um1 + sfw[c * 3 + 2] * u0 + sfb[c];
    }
    long long o = ((long long)(b * 128 + d)) * 2048 + t;
    x0a[o] = uc[0];
    vx[o]  = uc[2] * uc[1];
}

// ---------------------------------------------------------------------------
// Hyena implicit filter MLP: k[d, t] for one branch. grid=2048(t), block=128.
// ---------------------------------------------------------------------------
__global__ __launch_bounds__(128)
void k_hfilter(const float* __restrict__ W0, const float* __restrict__ b0, const float* __restrict__ fr0,
               const float* __restrict__ W1, const float* __restrict__ b1, const float* __restrict__ fr1,
               const float* __restrict__ W2, const float* __restrict__ b2, const float* __restrict__ fr2,
               const float* __restrict__ W3, float* __restrict__ kfilt)
{
    __shared__ float h0[64], h1[64], h2[64];
    const int t = blockIdx.x, tid = threadIdx.x;
    const float tval = (float)t / 4095.0f;
    const float w = 6.283185307179586f * (float)t / 4096.0f;
    const float z1 = cosf(1e-4f * w), z2 = -sinf(1e-4f * w);

    if (tid < 64) {
        float a = tval * W0[tid] + z1 * W0[64 + tid] + z2 * W0[128 + tid] + b0[tid];
        h0[tid] = sinf(fr0[tid] * a);
    }
    __syncthreads();
    if (tid < 64) {
        float a = b1[tid];
        for (int m = 0; m < 64; ++m) a += h0[m] * W1[m * 64 + tid];
        h1[tid] = sinf(fr1[tid] * a);
    }
    __syncthreads();
    if (tid < 64) {
        float a = b2[tid];
        for (int m = 0; m < 64; ++m) a += h1[m] * W2[m * 64 + tid];
        h2[tid] = sinf(fr2[tid] * a);
    }
    __syncthreads();
    {
        const int d = tid;
        float a = 0.f;
        for (int jj = 0; jj < 64; ++jj) a += h2[jj] * W3[jj * 128 + d];
        const float mind = -4.6051701859880914f / 1.5f;   // ln(1e-2)/1.5
        const float maxd = -4.6051701859880914f / 0.3f;   // ln(1e-2)/0.3
        float delta = fabsf(mind + (maxd - mind) * (float)d / 127.0f);
        kfilt[(long long)d * 2048 + t] = a * __expf(-tval * delta);
    }
}

// ---------------------------------------------------------------------------
// Stockham radix-2 FFT, N=4096 (in LDS, ping-pong). sign=+1 fwd, -1 inv(unnorm)
// ---------------------------------------------------------------------------
__device__ __forceinline__ void fft4096(float2* A, float2* B, int tid, float sign)
{
    float2* src = A; float2* dst = B;
    for (int Ns = 1; Ns < 4096; Ns <<= 1) {
        __syncthreads();
        for (int t = tid; t < 2048; t += 256) {
            int k = t & (Ns - 1);
            int j = ((t - k) << 1) + k;
            float ang = sign * (-6.283185307179586f) * ((float)k / (float)(Ns << 1));
            float sn, cs; __sincosf(ang, &sn, &cs);
            float2 x0 = src[t], x1 = src[t + 2048];
            float2 wv; wv.x = x1.x * cs - x1.y * sn; wv.y = x1.x * sn + x1.y * cs;
            dst[j].x      = x0.x + wv.x; dst[j].y      = x0.y + wv.y;
            dst[j + Ns].x = x0.x - wv.x; dst[j + Ns].y = x0.y - wv.y;
        }
        float2* tmp = src; src = dst; dst = tmp;
    }
    __syncthreads();   // 12 stages -> result back in A
}

// FFT of filter per channel d: kfc[d, 0..4095]. grid=128, block=256, dynLDS=64KB
__global__ __launch_bounds__(256)
void k_fftfilt(const float* __restrict__ kfilt, float2* __restrict__ kfc)
{
    extern __shared__ float2 fsm[];
    float2* A = fsm; float2* B = fsm + 4096;
    const int d = blockIdx.x, tid = threadIdx.x;
    for (int t = tid; t < 2048; t += 256) {
        A[t]        = make_float2(kfilt[(long long)d * 2048 + t], 0.f);
        A[t + 2048] = make_float2(0.f, 0.f);
    }
    fft4096(A, B, tid, 1.0f);
    for (int t = tid; t < 4096; t += 256) kfc[(long long)d * 4096 + t] = A[t];
}

// Causal FFT conv + D skip + x0 gate; writes hact[b*2048+t, d] ([16384,128]).
// grid = 1024 (b*128+d), block=256, dynLDS=64KB
__global__ __launch_bounds__(256)
void k_fftconv(const float* __restrict__ vx, const float* __restrict__ x0a,
               const float2* __restrict__ kfc, const float* __restrict__ Dp,
               float* __restrict__ hact)
{
    extern __shared__ float2 fsm[];
    float2* A = fsm; float2* B = fsm + 4096;
    const int bd = blockIdx.x, tid = threadIdx.x;
    const int b = bd >> 7, d = bd & 127;
    const long long base = (long long)bd * 2048;

    for (int t = tid; t < 2048; t += 256) {
        A[t]        = make_float2(vx[base + t], 0.f);
        A[t + 2048] = make_float2(0.f, 0.f);
    }
    fft4096(A, B, tid, 1.0f);
    for (int t = tid; t < 4096; t += 256) {
        float2 X = A[t], Kf = kfc[(long long)d * 4096 + t];
        A[t] = make_float2(X.x * Kf.x - X.y * Kf.y, X.x * Kf.y + X.y * Kf.x);
    }
    fft4096(A, B, tid, -1.0f);
    const float Dv = Dp[d];
    for (int t = tid; t < 2048; t += 256) {
        float y = A[t].x * (1.0f / 4096.0f) + vx[base + t] * Dv;
        hact[((long long)(b * 2048 + t)) * 128 + d] = y * x0a[base + t];
    }
}

// final: out[m] = sum_j t64[m,j]*W2[j] + b2
__global__ __launch_bounds__(256)
void k_finalvec(const float* __restrict__ t64, const float* __restrict__ W2,
                const float* __restrict__ b2, float* __restrict__ out)
{
    const long long m = (long long)blockIdx.x * 256 + threadIdx.x;
    float a = 0.f;
    for (int j = 0; j < 64; ++j) a += t64[m * 64 + j] * W2[j];
    out[m] = a + b2[0];
}

// ---------------------------------------------------------------------------
// Host orchestration
// ---------------------------------------------------------------------------
extern "C" void kernel_launch(void* const* d_in, const int* in_sizes, int n_in,
                              void* d_out, int out_size, void* d_ws, size_t ws_size,
                              hipStream_t stream)
{
    const int M = 16384;   // 8 * 2048
    // input order: z, propagate_pos, input_pos, then params flattened
    // (params as a jax pytree: keys sorted ASCII-alphabetically, nested dicts inline)
    const float* z_in = (const float*)d_in[0];
    const float* ppos = (const float*)d_in[1];
    const float* ipos = (const float*)d_in[2];
    const int PB = 3;
    auto P = [&](int k) -> const float* { return (const float*)d_in[PB + k]; };
    // sorted params index map
    const int iWk = 0, iWo = 1, iWq = 2, iWv = 3, ibo = 4, icpW = 5;
    auto ifW1 = [](int i){ return 6 + (i - 1) * 4 + 0; };
    auto ifW2 = [](int i){ return 6 + (i - 1) * 4 + 1; };
    auto ifb1 = [](int i){ return 6 + (i - 1) * 4 + 2; };
    auto ifb2 = [](int i){ return 6 + (i - 1) * 4 + 3; };
    const int iffnW1 = 22, iffnW2 = 23, iffnb1 = 24, iffnb2 = 25, igfft = 26;
    auto H = [](int i, int k){ return 27 + (i - 1) * 17 + k; };
    // within-h order: D, f_W0..f_W3, f_b0..f_b2, f_fr0..f_fr2, in_W, in_b, out_W, out_b, sf_b, sf_w
    const int hD=0,hfW0=1,hfW1=2,hfW2=3,hfW3=4,hfb0=5,hfb1=6,hfb2=7,hfr0=8,hfr1=9,hfr2=10,
              hinW=11,hinb=12,houtW=13,houtb=14,hsfb=15,hsfw=16;
    const int iln1b=95, iln1g=96, iln2b=97, iln2g=98, inb=99, ing=100,
              ioutW1=101, ioutW2=102, ioutb2=103;
    auto iprop = [](int j, int m){ return 104 + j * 3 + m; };

    // ---- workspace bump allocator ----
    char* wbase = (char*)d_ws;
    size_t woff = 0;
    auto alloc = [&](size_t bytes) -> void* {
        void* p = wbase + woff;
        woff = (woff + bytes + 255) & ~(size_t)255;
        return p;
    };
    float*     feats = (float*)alloc((size_t)M * 256 * 4);
    float*     x     = (float*)alloc((size_t)M * 128 * 4);
    float*     zn    = (float*)alloc((size_t)M * 128 * 4);
    float*     qf    = (float*)alloc((size_t)M * 1024 * 4);
    float*     kf    = (float*)alloc((size_t)M * 1024 * 4);
    float*     vf    = (float*)alloc((size_t)M * 1024 * 4);
    _Float16*  vh    = (_Float16*)alloc((size_t)M * 1024 * 2);
    _Float16*  qr    = (_Float16*)alloc((size_t)M * 1024 * 2);
    _Float16*  kr    = (_Float16*)alloc((size_t)M * 1024 * 2);
    float*     dots  = (float*)alloc((size_t)64 * 128 * 128 * 4);
    _Float16*  dotsh = (_Float16*)alloc((size_t)64 * 128 * 128 * 2);
    float*     xattn = (float*)alloc((size_t)M * 1024 * 4);
    float*     attno = (float*)alloc((size_t)M * 128 * 4);
    float*     xa    = (float*)alloc((size_t)M * 128 * 4);
    float*     ffn1  = (float*)alloc((size_t)M * 128 * 4);
    float*     xb    = (float*)alloc((size_t)M * 128 * 4);
    float*     zn2   = (float*)alloc((size_t)M * 128 * 4);
    float*     accb  = (float*)alloc((size_t)M * 128 * 4);
    float*     u     = (float*)alloc((size_t)M * 384 * 4);
    float*     vx    = (float*)alloc((size_t)8 * 128 * 2048 * 4);
    float*     x0a   = (float*)alloc((size_t)8 * 128 * 2048 * 4);
    float*     kfilt = (float*)alloc((size_t)128 * 2048 * 4);
    float2*    kfc   = (float2*)alloc((size_t)128 * 4096 * 8);
    float*     hact  = (float*)alloc((size_t)M * 128 * 4);
    float*     hout  = (float*)alloc((size_t)M * 128 * 4);
    float*     xi    = (float*)alloc((size_t)M * 128 * 4);
    float*     tb    = (float*)alloc((size_t)M * 256 * 4);
    float*     p1    = (float*)alloc((size_t)M * 128 * 4);
    float*     p2    = (float*)alloc((size_t)M * 128 * 4);
    float*     t64   = (float*)alloc((size_t)M * 64 * 4);
    (void)in_sizes; (void)n_in; (void)out_size; (void)ws_size;

    auto g = [](int N, int Mm, int B){ return dim3((N + 127) / 128, (Mm + 127) / 128, B); };

    // ---- stage A: Fourier features + coordinate projection ----
    k_feats<<<dim3(M), 256, 0, stream>>>(ppos, P(igfft), feats);
    k_gemm<false,false,false,0,false,false><<<g(128, M, 1), 256, 0, stream>>>(
        feats, P(icpW), nullptr, x, nullptr, nullptr,
        M, 128, 256, 256, 128, 128, 1.0f, 1, 0,0, 0,0, 0,0);

    // ---- stage B: zn = LN(z) ----
    k_layernorm<false><<<dim3(M), 128, 0, stream>>>(z_in, P(iln1g), P(iln1b), nullptr, zn);

    // ---- stage C: q, k, v projections ----
    k_gemm<false,false,false,0,false,false><<<g(1024, M, 1), 256, 0, stream>>>(
        x, P(iWq), nullptr, qf, nullptr, nullptr, M, 1024, 128, 128, 1024, 1024, 1.0f, 1, 0,0, 0,0, 0,0);
    k_gemm<false,false,false,0,false,false><<<g(1024, M, 1), 256, 0, stream>>>(
        zn, P(iWk), nullptr, kf, nullptr, nullptr, M, 1024, 128, 128, 1024, 1024, 1.0f, 1, 0,0, 0,0, 0,0);
    k_gemm<false,false,false,0,false,true><<<g(1024, M, 1), 256, 0, stream>>>(
        zn, P(iWv), nullptr, vf, vh, nullptr, M, 1024, 128, 128, 1024, 1024, 1.0f, 1, 0,0, 0,0, 0,0);

    // ---- stage D: instance-norm + rotary -> f16 fragments ----
    k_instrot<<<dim3(64, 8, 8), 256, 0, stream>>>(qf, ppos, qr);
    k_instrot<<<dim3(64, 8, 8), 256, 0, stream>>>(kf, ipos, kr);

    // ---- stage E: dots = k^T v per (b,h); out = q @ dots / n ----
    k_gemm<true,true,true,0,false,true><<<dim3(1, 1, 64), 256, 0, stream>>>(
        kr, vh, nullptr, dots, dotsh, nullptr,
        128, 128, 2048, 128, 1024, 128, 1.0f, 8,
        (long long)8*2048*128, (long long)2048*128,
        (long long)2048*1024, 128,
        (long long)8*128*128, (long long)128*128);
    k_gemm<false,true,true,0,false,false><<<dim3(1, 16, 64), 256, 0, stream>>>(
        qr, dotsh, nullptr, xattn, nullptr, nullptr,
        2048, 128, 128, 128, 128, 1024, 1.0f / 2048.0f, 8,
        (long long)8*2048*128, (long long)2048*128,
        (long long)8*16384, 16384,
        (long long)2048*1024, 128);

    // ---- stage F: out proj + LN residual + FFN ----
    k_gemm<false,false,false,0,false,false><<<g(128, M, 1), 256, 0, stream>>>(
        xattn, P(iWo), P(ibo), attno, nullptr, nullptr, M, 128, 1024, 1024, 128, 128, 1.0f, 1, 0,0, 0,0, 0,0);
    k_layernorm<true><<<dim3(M), 128, 0, stream>>>(attno, P(iln2g), P(iln2b), x, xa);
    k_gemm<false,false,false,1,false,false><<<g(128, M, 1), 256, 0, stream>>>(
        xa, P(iffnW1), P(iffnb1), ffn1, nullptr, nullptr, M, 128, 128, 128, 128, 128, 1.0f, 1, 0,0, 0,0, 0,0);
    k_gemm<false,false,false,0,true,false><<<g(128, M, 1), 256, 0, stream>>>(
        ffn1, P(iffnW2), P(iffnb2), xb, nullptr, xa, M, 128, 128, 128, 128, 128, 1.0f, 1, 0,0, 0,0, 0,0);

    // ---- stage G: 4 Hyena branches (shared zn2) ----
    k_layernorm<false><<<dim3(M), 128, 0, stream>>>(xb, P(ing), P(inb), nullptr, zn2);
    for (int br = 1; br <= 4; ++br) {
        k_gemm<false,false,false,0,false,false><<<g(384, M, 1), 256, 0, stream>>>(
            zn2, P(H(br,hinW)), P(H(br,hinb)), u, nullptr, nullptr,
            M, 384, 128, 128, 384, 384, 1.0f, 1, 0,0, 0,0, 0,0);
        k_shortconv<<<dim3(2048, 8), 128, 0, stream>>>(u, P(H(br,hsfw)), P(H(br,hsfb)), vx, x0a);
        k_hfilter<<<dim3(2048), 128, 0, stream>>>(
            P(H(br,hfW0)), P(H(br,hfb0)), P(H(br,hfr0)),
            P(H(br,hfW1)), P(H(br,hfb1)), P(H(br,hfr1)),
            P(H(br,hfW2)), P(H(br,hfb2)), P(H(br,hfr2)),
            P(H(br,hfW3)), kfilt);
        k_fftfilt<<<dim3(128), 256, 65536, stream>>>(kfilt, kfc);
        k_fftconv<<<dim3(1024), 256, 65536, stream>>>(vx, x0a, kfc, P(H(br,hD)), hact);
        k_gemm<false,false,false,0,false,false><<<g(128, M, 1), 256, 0, stream>>>(
            hact, P(H(br,houtW)), P(H(br,houtb)), hout, nullptr, nullptr,
            M, 128, 128, 128, 128, 128, 1.0f, 1, 0,0, 0,0, 0,0);
        k_layernorm<true><<<dim3(M), 128, 0, stream>>>(hout, P(ing), P(inb), xb, xi);
        k_gemm<false,false,false,1,false,false><<<g(256, M, 1), 256, 0, stream>>>(
            xi, P(ifW1(br)), P(ifb1(br)), tb, nullptr, nullptr,
            M, 256, 128, 128, 256, 256, 1.0f, 1, 0,0, 0,0, 0,0);
        if (br == 1)
            k_gemm<false,false,false,0,false,false><<<g(128, M, 1), 256, 0, stream>>>(
                tb, P(ifW2(br)), P(ifb2(br)), accb, nullptr, nullptr,
                M, 128, 256, 256, 128, 128, 1.0f, 1, 0,0, 0,0, 0,0);
        else
            k_gemm<false,false,false,0,true,false><<<g(128, M, 1), 256, 0, stream>>>(
                tb, P(ifW2(br)), P(ifb2(br)), accb, nullptr, accb,
                M, 128, 256, 256, 128, 128, 1.0f, 1, 0,0, 0,0, 0,0);
    }

    // ---- stage H: propagator residual MLP blocks ----
    for (int j = 0; j < 2; ++j) {
        k_gemm<false,false,false,1,false,false><<<g(128, M, 1), 256, 0, stream>>>(
            accb, P(iprop(j,0)), nullptr, p1, nullptr, nullptr, M, 128, 128, 128, 128, 128, 1.0f, 1, 0,0, 0,0, 0,0);
        k_gemm<false,false,false,1,false,false><<<g(128, M, 1), 256, 0, stream>>>(
            p1, P(iprop(j,1)), nullptr, p2, nullptr, nullptr, M, 128, 128, 128, 128, 128, 1.0f, 1, 0,0, 0,0, 0,0);
        k_gemm<false,false,false,0,true,false><<<g(128, M, 1), 256, 0, stream>>>(
            p2, P(iprop(j,2)), nullptr, accb, nullptr, accb, M, 128, 128, 128, 128, 128, 1.0f, 1, 0,0, 0,0, 0,0);
    }

    // ---- stage I: output head ----
    k_gemm<false,false,false,1,false,false><<<g(64, M, 1), 256, 0, stream>>>(
        accb, P(ioutW1), nullptr, t64, nullptr, nullptr, M, 64, 128, 128, 64, 64, 1.0f, 1, 0,0, 0,0, 0,0);
    k_finalvec<<<dim3(M / 256), 256, 0, stream>>>(t64, P(ioutW2), P(ioutb2), (float*)d_out);
}
